// LearnedIndexer_65687229825517
// MI455X (gfx1250) — matrix operand order repaired and checked by reading, not compiled, not run on hardware
//
#include <hip/hip_runtime.h>
#include <hip/hip_bf16.h>
#include <stdint.h>

typedef __attribute__((ext_vector_type(16))) __bf16   v16bf;
typedef __attribute__((ext_vector_type(8)))  float    v8f;
typedef __attribute__((ext_vector_type(4)))  float    v4f;
typedef __attribute__((ext_vector_type(4)))  unsigned v4u;
typedef __attribute__((ext_vector_type(8)))  int      v8i;
typedef __attribute__((ext_vector_type(4)))  int      v4i;

#define HEADS 16
#define TOK   4096
#define HD    128
#define DM    2048   /* HEADS*HD  */
#define PO    1024   /* HEADS*IDX */
#define IDXD  64
#define TOPK  128

__device__ __forceinline__ __bf16 f2bf(float f) {
  union { float f; unsigned u; } v; v.f = f;
  unsigned r = v.u + 0x7FFFu + ((v.u >> 16) & 1u);   // round-to-nearest-even
  unsigned short h = (unsigned short)(r >> 16);
  __bf16 b; __builtin_memcpy(&b, &h, 2);
  return b;
}

// monotone float -> uint key (total order, ascending)
__device__ __forceinline__ unsigned ordkey(float f) {
  union { float f; unsigned u; } v; v.f = f;
  return (v.u & 0x80000000u) ? ~v.u : (v.u | 0x80000000u);
}

// ---------------------------------------------------------------------------
// Kernel 0: W (DM x PO fp32, row-major) -> Wt (PO x DM bf16, row-major)
// Tiled 32x32 transpose through LDS; makes WMMA B-fragments contiguous.
// ---------------------------------------------------------------------------
__global__ __launch_bounds__(256) void wtrans_kernel(
    const float* __restrict__ W, __bf16* __restrict__ Wt)
{
  __shared__ float tile[32][33];
  const int cT = blockIdx.x * 32;   // model-dim tile origin
  const int nT = blockIdx.y * 32;   // proj-dim tile origin
  const int tx = threadIdx.x;       // 0..31
  const int ty = threadIdx.y;       // 0..7
  for (int yy = 0; yy < 32; yy += 8)
    tile[ty + yy][tx] = W[(size_t)(cT + ty + yy) * PO + nT + tx];
  __syncthreads();
  for (int yy = 0; yy < 32; yy += 8)
    Wt[(size_t)(nT + ty + yy) * DM + cT + tx] = f2bf(tile[tx][ty + yy]);
}

// ---------------------------------------------------------------------------
// Kernel 1: low-rank projection GEMM  X_idx = X_flat @ W  (bf16 in, f32 acc)
//   x : (H, TOK, HD) fp32 ; logical A row t, col c -> x[c/HD][t][c%HD]
//   Wt: (PO, DM) bf16 (transposed weights -> contiguous B fragments)
//   out: (H, TOK, IDXD) bf16; N-tile of 64 == one head
// A panel (128 x 32 fp32 2D tile, row stride HD) staged to LDS by the TDM.
// WG: 256 thr (8 waves). Tile 128x64, K-step 32.
// ---------------------------------------------------------------------------
__global__ __launch_bounds__(256) void proj_kernel(
    const float* __restrict__ x,
    const __bf16* __restrict__ Wt,
    __bf16* __restrict__ xidx)
{
  __shared__ float Asf[128][32];          // 16 KB TDM destination

  const int mTile = blockIdx.x;           // 0..31
  const int h     = blockIdx.y;           // 0..15
  const int tid   = threadIdx.x;
  const int wave  = tid >> 5;
  const int lane  = tid & 31;
  const int lan16 = lane & 15;
  const int rbase = (lane < 16) ? 0 : 8;  // D row offset per lane half
  const int koffA = (lane < 16) ? 0 : 16; // A-frag K offset per lane half
  const int koffB = (lane < 16) ? 0 : 16; // B-frag K offset per lane half

  const unsigned lds_base = (unsigned)(uintptr_t)(&Asf[0][0]);

  v8f c[4] = {};

  for (int kt = 0; kt < DM; kt += 32) {
    __syncthreads();   // previous tile fully consumed
    if (tid < 32) {
      // --- Tensor DMA: 2D tile 32(d) x 128(t) fp32, row stride HD elems ---
      const int hh = kt >> 7, d0 = kt & 127;
      const float* gsrc = x + ((size_t)hh * TOK + (size_t)mTile * 128) * HD + d0;
      unsigned long long ga = (unsigned long long)(uintptr_t)gsrc;
      v4u g0;                                   // D# group 0
      g0[0] = 1u;                               // count=1 (valid descriptor)
      g0[1] = lds_base;                         // lds_addr
      g0[2] = (unsigned)ga;                     // global_addr[31:0]
      g0[3] = (unsigned)((ga >> 32) & 0x01FFFFFFull) | (2u << 30); // addr[56:32] | type=2
      v8i g1;                                   // D# group 1
      g1[0] = 0x00020000;                       // wg_mask=0, data_size=4B
      g1[1] = (int)(32u  << 16);                // tensor_dim0 = 32
      g1[2] = (int)(128u << 16);                // tensor_dim1 = 128
      g1[3] = (int)(32u  << 16);                // tile_dim0   = 32
      g1[4] = 128;                              // tile_dim1   = 128, tile_dim2=0
      g1[5] = 128;                              // tensor_dim0_stride = HD = 128
      g1[6] = 0;
      g1[7] = 0;
      v4i gz = {0, 0, 0, 0};                    // groups 2/3 unused (2D tensor)
#if defined(__clang_major__) && __clang_major__ >= 23
      v8i gz8 = {0, 0, 0, 0, 0, 0, 0, 0};
      __builtin_amdgcn_tensor_load_to_lds(g0, g1, gz, gz, gz8, 0);
#else
      __builtin_amdgcn_tensor_load_to_lds(g0, g1, gz, gz, 0);
#endif
      __builtin_amdgcn_s_wait_tensorcnt(0);
    }
    __syncthreads();   // tile visible to all waves

    // A fragment (16x32 bf16): lanes 0-15: M=lane, K {0..15};
    //                          lanes16-31: M=lane-16, K {16..31}
    const int m = wave * 16 + lan16;
    const v4f* rowp = (const v4f*)&Asf[m][0];
    v4f f0 = rowp[(koffA >> 2) + 0];
    v4f f1 = rowp[(koffA >> 2) + 1];
    v4f f2 = rowp[(koffA >> 2) + 2];
    v4f f3 = rowp[(koffA >> 2) + 3];
    v16bf a;
    for (int e = 0; e < 4; ++e) {
      a[e]      = f2bf(f0[e]);
      a[4 + e]  = f2bf(f1[e]);
      a[8 + e]  = f2bf(f2[e]);
      a[12 + e] = f2bf(f3[e]);
    }
    // B fragments: contiguous 16-elem (32B) global loads from transposed W
    for (int cb = 0; cb < 4; ++cb) {
      int colb = cb * 16 + lan16;
      v16bf b = *(const v16bf*)(Wt + (size_t)(h * 64 + colb) * DM + kt + koffB);
      c[cb] = __builtin_amdgcn_wmma_f32_16x16x32_bf16(
          false, a, false, b, (short)0, c[cb], false, false);
    }
  }

  // D layout: VGPR r -> M = r + rbase, N = lan16
  for (int cb = 0; cb < 4; ++cb)
    for (int r = 0; r < 8; ++r) {
      int t = mTile * 128 + wave * 16 + rbase + r;
      int j = cb * 16 + lan16;
      xidx[((size_t)h * TOK + t) * IDXD + j] = f2bf(c[cb][r]);
    }
}

// ---------------------------------------------------------------------------
// Kernel 2: fused scores (WMMA) + exact per-row top-128 (radix select + sort)
// WG = 512 thr (16 waves) per (head, 16 q-rows). Scores live only in LDS.
// ---------------------------------------------------------------------------
#define SC_BYTES   (16 * 4096 * 4)          /* 262144 */
#define HI_OFF     SC_BYTES
#define HI_BYTES   (16 * 256 * 4)           /* 16384  */
#define ST_OFF     (HI_OFF + HI_BYTES)      /* 278528 */
#define ST_BYTES   (16 * 128 * 8)           /* 16384  */
#define MISC_OFF   (ST_OFF + ST_BYTES)      /* 294912 */
#define SMEM_TOTAL (MISC_OFF + 4 * 16 * 4)  /* 295168 */

__global__ __launch_bounds__(512) void score_topk_kernel(
    const __bf16* __restrict__ qidx,
    const __bf16* __restrict__ kidx,
    int* __restrict__ out)
{
  extern __shared__ char smem[];
  float*              scores = (float*)smem;                         // [16][4096]
  unsigned*           hist   = (unsigned*)(smem + HI_OFF);           // [16][256]
  unsigned long long* stage  = (unsigned long long*)(smem + ST_OFF); // [16][128]
  unsigned* selA = (unsigned*)(smem + MISC_OFF);
  unsigned* selB = selA + 16;
  unsigned* cntA = selB + 16;
  unsigned* cntB = cntA + 16;

  const int qt    = blockIdx.x;          // 0..255  (16 q rows per WG)
  const int h     = blockIdx.y;          // 0..15
  const int tid   = threadIdx.x;
  const int w     = tid >> 5;            // wave id 0..15
  const int lane  = tid & 31;
  const int lan16 = lane & 15;
  const int rbase = (lane < 16) ? 0 : 8;
  const int koffA = (lane < 16) ? 0 : 16;
  const int koffB = (lane < 16) ? 0 : 16;

  // ---- phase 1: 16x4096 score strip via bf16 WMMA (K=64 as 2x32) ----
  const __bf16* qp = qidx + ((size_t)h * TOK + qt * 16 + lan16) * IDXD;
  v16bf a0, a1;
  for (int e = 0; e < 8; ++e) {
    a0[e]     = qp[koffA + e];
    a0[8 + e] = qp[8 + koffA + e];
    a1[e]     = qp[32 + koffA + e];
    a1[8 + e] = qp[40 + koffA + e];
  }
  for (int kt = w; kt < 256; kt += 16) {
    const __bf16* kp = kidx + ((size_t)h * TOK + kt * 16 + lan16) * IDXD;
    v16bf b0 = *(const v16bf*)(kp + koffB);
    v16bf b1 = *(const v16bf*)(kp + 32 + koffB);
    v8f c = {};
    c = __builtin_amdgcn_wmma_f32_16x16x32_bf16(false, a0, false, b0, (short)0, c, false, false);
    c = __builtin_amdgcn_wmma_f32_16x16x32_bf16(false, a1, false, b1, (short)0, c, false, false);
    int colg = kt * 16 + lan16;
    for (int r = 0; r < 8; ++r) scores[(rbase + r) * 4096 + colg] = c[r];
  }
  __syncthreads();

  // ---- phase 2: exact radix-select of 128th-largest key (row = wave) ----
  unsigned prefix = 0, want = TOPK;
  for (int round = 0; round < 4; ++round) {
    int shift = 24 - round * 8;
    for (int i = lane; i < 256; i += 32) hist[w * 256 + i] = 0u;
    __syncthreads();
    unsigned hmask = (round == 0) ? 0u : (0xFFFFFFFFu << (shift + 8));
    for (int i = lane; i < 4096; i += 32) {
      unsigned key = ordkey(scores[w * 4096 + i]);
      if ((key & hmask) == prefix)
        atomicAdd(&hist[w * 256 + ((key >> shift) & 255u)], 1u);
    }
    __syncthreads();
    if (lane == 0) {
      unsigned cum = 0; int d = 255;
      for (; d > 0; --d) {
        unsigned cc = hist[w * 256 + d];
        if (cum + cc >= want) break;
        cum += cc;
      }
      selA[w] = prefix | ((unsigned)d << shift);
      selB[w] = want - cum;
    }
    __syncthreads();
    prefix = selA[w];
    want   = selB[w];
    __syncthreads();
  }
  const unsigned thr = prefix;   // key of the 128th largest element

  // ---- phase 3: compaction (strictly-greater, then equals up to `want`) ----
  if (lane == 0) { cntA[w] = 0u; cntB[w] = 0u; }
  __syncthreads();
  for (int i = lane; i < 4096; i += 32) {
    unsigned key = ordkey(scores[w * 4096 + i]);
    if (key > thr) {
      unsigned p = atomicAdd(&cntA[w], 1u);
      stage[w * 128 + p] =
          ((unsigned long long)key << 32) | (unsigned)(0xFFFFFFFFu - (unsigned)i);
    }
  }
  __syncthreads();
  const unsigned nGT = cntA[w];
  for (int i = lane; i < 4096; i += 32) {
    unsigned key = ordkey(scores[w * 4096 + i]);
    if (key == thr) {
      unsigned p = atomicAdd(&cntB[w], 1u);
      if (p < want)
        stage[w * 128 + nGT + p] =
            ((unsigned long long)key << 32) | (unsigned)(0xFFFFFFFFu - (unsigned)i);
    }
  }
  __syncthreads();

  // ---- phase 4: bitonic sort 128 composites descending ----
  // composite = (key<<32) | (~idx)  ->  value desc, index asc on ties
  for (int k2 = 2; k2 <= 128; k2 <<= 1) {
    for (int j2 = k2 >> 1; j2 > 0; j2 >>= 1) {
      for (int idx = lane; idx < 128; idx += 32) {
        int l = idx ^ j2;
        if (l > idx) {
          unsigned long long A = stage[w * 128 + idx];
          unsigned long long B = stage[w * 128 + l];
          bool desc = ((idx & k2) == 0);
          bool swp  = desc ? (A < B) : (A > B);
          if (swp) { stage[w * 128 + idx] = B; stage[w * 128 + l] = A; }
        }
      }
      __syncthreads();
    }
  }

  // ---- output: indices (B,H,T_q,TOPK) int32 ----
  int qrow = qt * 16 + w;
  for (int i = lane; i < 128; i += 32)
    out[((size_t)h * TOK + qrow) * TOPK + i] =
        (int)(0xFFFFFFFFu - (unsigned)(stage[w * 128 + i] & 0xFFFFFFFFull));
}

// ---------------------------------------------------------------------------
extern "C" void kernel_launch(void* const* d_in, const int* in_sizes, int n_in,
                              void* d_out, int out_size, void* d_ws, size_t ws_size,
                              hipStream_t stream) {
  const float* q  = (const float*)d_in[0];
  const float* k  = (const float*)d_in[1];
  const float* Wq = (const float*)d_in[2];
  const float* Wk = (const float*)d_in[3];
  (void)in_sizes; (void)n_in; (void)out_size; (void)ws_size;

  __bf16* qidx = (__bf16*)d_ws;                                  // 8 MB
  __bf16* kidx = qidx + (size_t)HEADS * TOK * IDXD;              // 8 MB
  __bf16* Wtq  = kidx + (size_t)HEADS * TOK * IDXD;              // 4 MB
  __bf16* Wtk  = Wtq  + (size_t)PO * DM;                         // 4 MB

  dim3 gt(DM / 32, PO / 32, 1);
  dim3 bt(32, 8, 1);
  wtrans_kernel<<<gt, bt, 0, stream>>>(Wq, Wtq);
  wtrans_kernel<<<gt, bt, 0, stream>>>(Wk, Wtk);

  dim3 g1(TOK / 128, HEADS, 1);
  proj_kernel<<<g1, 256, 0, stream>>>(q, Wtq, qidx);
  proj_kernel<<<g1, 256, 0, stream>>>(k, Wtk, kidx);

  dim3 g2(TOK / 16, HEADS, 1);
  score_topk_kernel<<<g2, 512, (size_t)SMEM_TOTAL, stream>>>(qidx, kidx, (int*)d_out);
}
